// CTGRU_1236950582111
// MI455X (gfx1250) — matrix operand order repaired and checked by reading, not compiled, not run on hardware
//
#include <hip/hip_runtime.h>
#include <hip/hip_bf16.h>
#include <stdint.h>

// Problem dims
#define B_  32
#define T_  512
#define I_  256
#define H_  1024
#define M_  8
#define O_  512
#define IH  1280   // I + H
#define MH  8192   // M * H

typedef __attribute__((ext_vector_type(16))) __bf16 v16bf;
typedef __attribute__((ext_vector_type(8)))  __bf16 bf16x8;
typedef __attribute__((ext_vector_type(8)))  float  v8f;

struct Pair16 { bf16x8 lo, hi; };

#define LN10_HALF 1.1512925464970229f   // ln(10)/2

static __device__ __forceinline__ unsigned short f2bf_bits(float f) {
  unsigned u = __builtin_bit_cast(unsigned, f);
  u += 0x7fffu + ((u >> 16) & 1u);               // round-to-nearest-even
  return (unsigned short)(u >> 16);
}
static __device__ __forceinline__ __bf16 tobf(float f) {
  unsigned short s = f2bf_bits(f);
  return __builtin_bit_cast(__bf16, s);
}
static __device__ __forceinline__ v16bf make_afrag(const __bf16* p0, const __bf16* p1) {
  Pair16 s;
  s.lo = *(const bf16x8*)p0;
  s.hi = *(const bf16x8*)p1;
  return __builtin_bit_cast(v16bf, s);
}
static __device__ __forceinline__ v8f wmma_bf16(v16bf a, v16bf b, v8f c) {
  return __builtin_amdgcn_wmma_f32_16x16x32_bf16(false, a, false, b, (short)0, c,
                                                 false, false);
}

// Stage `nbytes` (multiple of 16*256) from global into LDS with the CDNA5
// async Global->LDS path; each lane moves 16B per instruction.
static __device__ __forceinline__ void stage_async(const __bf16* __restrict__ g,
                                                   __bf16* smem, int nbytes) {
  unsigned lbase = (unsigned)(uintptr_t)smem;       // low 32b of generic = LDS offset
  int nchunk = nbytes >> 4;
  for (int c = threadIdx.x; c < nchunk; c += 256) {
    unsigned laddr = lbase + (unsigned)(c << 4);
    unsigned long long ga = (unsigned long long)(uintptr_t)(g + (c << 3));
    asm volatile("global_load_async_to_lds_b128 %0, %1, off"
                 :: "v"(laddr), "v"(ga) : "memory");
  }
  asm volatile("s_wait_asynccnt 0x0" ::: "memory");
  __syncthreads();
}

// ---------------------------------------------------------------------------
// Pack fp32 weight [K][N] row-major into per-wave B-fragment tiles (bf16).
// ---------------------------------------------------------------------------
__global__ void pack_weight(const float* __restrict__ W, unsigned* __restrict__ out,
                            int K, int N, int KT) {
  int tid = blockIdx.x * blockDim.x + threadIdx.x;
  int total = (K * N) >> 1;
  if (tid >= total) return;
  int v    = tid & 7;
  int lane = (tid >> 3) & 31;
  int tile = tid >> 8;
  int kt   = tile % KT;
  int nt   = tile / KT;
  int n    = nt * 16 + (lane & 15);
  int kb   = kt * 32 + ((v >> 2) << 4) + ((lane >> 4) << 3) + ((v & 3) << 1);
  unsigned lo = f2bf_bits(W[(size_t)kb * N + n]);
  unsigned hi = f2bf_bits(W[(size_t)(kb + 1) * N + n]);
  out[tid] = lo | (hi << 16);
}

// ---------------------------------------------------------------------------
// Zero h_hat; fused_0 = [x_0 | 0] bf16 row-major [32][1280]
// ---------------------------------------------------------------------------
__global__ void init_state(const float* __restrict__ x, float* __restrict__ hhat,
                           __bf16* __restrict__ fused) {
  int idx = blockIdx.x * blockDim.x + threadIdx.x;
  if (idx < B_ * MH) hhat[idx] = 0.0f;
  if (idx < B_ * IH) {
    int b = idx / IH, c = idx % IH;
    fused[idx] = (c < I_) ? tobf(x[((size_t)b * T_) * I_ + c]) : tobf(0.0f);
  }
}

// ---------------------------------------------------------------------------
// Gates GEMM + fused softmax epilogues.
//  sel=0: lnr = fused@Wr + br; rki = softmax_m(-(lnr-ln_tau)^2);
//         q = sum_m rki*hhat  ->  xq[:, 256+h] (bf16)
//  sel=1: lns = fused@Wu + bu; ski = softmax  ->  ski[32][8192] (fp32)
// Also copies x_t -> xq[:, :256].  1024 waves -> <<<128, 256>>>
// ---------------------------------------------------------------------------
__global__ void gates_gemm(const __bf16* __restrict__ fused,
                           const __bf16* __restrict__ Wrp, const __bf16* __restrict__ Wup,
                           const float* __restrict__ br,  const float* __restrict__ bu,
                           const float* __restrict__ hhat, const float* __restrict__ x,
                           int step, __bf16* __restrict__ xq, float* __restrict__ ski) {
  __shared__ __bf16 smem[B_ * IH];                  // 80 KB activation stage
  int gtid = blockIdx.x * 256 + threadIdx.x;

  // x_t -> xq x-part (xq is only read by decay_gemm, later in the stream)
  if (gtid < B_ * I_) {
    int b = gtid >> 8, c = gtid & 255;
    xq[b * IH + c] = tobf(x[((size_t)b * T_ + step) * I_ + c]);
  }

  stage_async(fused, smem, B_ * IH * 2);

  int wave = gtid >> 5;
  int lane = gtid & 31;
  int sel  = wave & 1;                              // wave-uniform
  int nt   = wave >> 1;                             // 0..511
  const __bf16* Wp   = sel ? Wup : Wrp;
  const float*  bias = sel ? bu  : br;

  int m0 = lane & 15;
  int hi = lane >> 4;
  const __bf16* a0r = smem + m0 * IH;
  const __bf16* a1r = smem + (m0 + 16) * IH;
  const __bf16* bp  = Wp + ((size_t)(nt * 40) * 32 + lane) * 16;

  v8f c0 = {0.f,0.f,0.f,0.f,0.f,0.f,0.f,0.f};
  v8f c1 = {0.f,0.f,0.f,0.f,0.f,0.f,0.f,0.f};
  for (int kt = 0; kt < 40; ++kt) {
    int k0 = kt * 32;
    v16bf bfrag = *(const v16bf*)bp;                // 32B/lane, coalesced
    bp += 32 * 16;
    __builtin_prefetch(bp, 0, 1);
    v16bf a0 = make_afrag(a0r + k0 + 8 * hi, a0r + k0 + 16 + 8 * hi);   // ds_load
    v16bf a1 = make_afrag(a1r + k0 + 8 * hi, a1r + k0 + 16 + 8 * hi);
    c0 = wmma_bf16(a0, bfrag, c0);
    c1 = wmma_bf16(a1, bfrag, c1);
  }

  int col = nt * 16 + m0;                           // lane's output column
  int m   = m0 & 7;                                 // tau index
  int h   = col >> 3;                               // h-unit
  float lt = LN10_HALF * (float)m;
  float bv = bias[col];

  if (sel == 0) {
    // rki softmax + q reduction across the 8-lane m-group, per batch row
    #pragma unroll
    for (int r = 0; r < 8; ++r) {
      #pragma unroll
      for (int half = 0; half < 2; ++half) {
        int   row = r + 8 * hi + 16 * half;
        float v   = (half ? c1[r] : c0[r]) + bv;
        float d   = v - lt;
        float s   = -d * d;
        float mx  = s;
        mx = fmaxf(mx, __shfl_xor(mx, 1, 32));
        mx = fmaxf(mx, __shfl_xor(mx, 2, 32));
        mx = fmaxf(mx, __shfl_xor(mx, 4, 32));
        float w   = __expf(s - mx);
        float hv  = hhat[(size_t)row * MH + col];
        float num = w * hv, den = w;
        num += __shfl_xor(num, 1, 32); den += __shfl_xor(den, 1, 32);
        num += __shfl_xor(num, 2, 32); den += __shfl_xor(den, 2, 32);
        num += __shfl_xor(num, 4, 32); den += __shfl_xor(den, 4, 32);
        if (m == 0) xq[row * IH + I_ + h] = tobf(num / den);
      }
    }
  } else {
    // ski softmax, written normalized
    #pragma unroll
    for (int r = 0; r < 8; ++r) {
      #pragma unroll
      for (int half = 0; half < 2; ++half) {
        int   row = r + 8 * hi + 16 * half;
        float v   = (half ? c1[r] : c0[r]) + bv;
        float d   = v - lt;
        float s   = -d * d;
        float mx  = s;
        mx = fmaxf(mx, __shfl_xor(mx, 1, 32));
        mx = fmaxf(mx, __shfl_xor(mx, 2, 32));
        mx = fmaxf(mx, __shfl_xor(mx, 4, 32));
        float w   = __expf(s - mx);
        float den = w;
        den += __shfl_xor(den, 1, 32);
        den += __shfl_xor(den, 2, 32);
        den += __shfl_xor(den, 4, 32);
        ski[(size_t)row * MH + col] = w / den;
      }
    }
  }
}

// ---------------------------------------------------------------------------
// Decay GEMM + full state update epilogue.
//  qk = tanh(xq@Wd + bd); hhat = ((1-ski)*hhat + ski*qk)*exp(-t_quirk/tau)
//  writes hhat, hflat(bf16), fused[:,256+h] = sum_m, and x_{t+1} -> fused x-part
// 64 waves -> <<<8, 256>>>
// ---------------------------------------------------------------------------
__global__ void decay_gemm(const __bf16* __restrict__ xq, const __bf16* __restrict__ Wdp,
                           const float* __restrict__ bd,  const float* __restrict__ ski,
                           const float* __restrict__ tarr, const float* __restrict__ x,
                           int step, float* __restrict__ hhat,
                           __bf16* __restrict__ fused, __bf16* __restrict__ hflat) {
  __shared__ __bf16 smem[B_ * IH];                  // 80 KB activation stage
  __shared__ float  ecol[B_ * M_];                  // exp(-t_quirk/tau) table
  int gtid = blockIdx.x * 256 + threadIdx.x;

  // x_{t+1} -> fused x-part (fused is only read by next step's gates_gemm)
  if (step + 1 < T_) {
    for (int j = gtid; j < B_ * I_; j += 8 * 256) {
      int b = j >> 8, c = j & 255;
      fused[b * IH + c] = tobf(x[((size_t)b * T_ + (step + 1)) * I_ + c]);
    }
  }
  // faithful torch quirk: exp_term[b,m] = exp(-t[(b*8+m)%32, step] / tau[m])
  {
    int tid = threadIdx.x;                          // tid = b*8+m, b<32, m<8
    int b = tid >> 3, m = tid & 7;
    float tv  = tarr[(size_t)((b * M_ + m) & (B_ - 1)) * T_ + step];
    float itau = __expf(-LN10_HALF * (float)m);     // 1/tau[m]
    ecol[tid] = __expf(-tv * itau);
  }

  stage_async(xq, smem, B_ * IH * 2);

  int wave = gtid >> 5;                             // 0..63 = n-tile
  int lane = gtid & 31;
  int nt = wave;
  int m0 = lane & 15;
  int hi = lane >> 4;
  const __bf16* a0r = smem + m0 * IH;
  const __bf16* a1r = smem + (m0 + 16) * IH;
  const __bf16* bp  = Wdp + ((size_t)(nt * 40) * 32 + lane) * 16;

  v8f c0 = {0.f,0.f,0.f,0.f,0.f,0.f,0.f,0.f};
  v8f c1 = {0.f,0.f,0.f,0.f,0.f,0.f,0.f,0.f};
  for (int kt = 0; kt < 40; ++kt) {
    int k0 = kt * 32;
    v16bf bfrag = *(const v16bf*)bp;
    bp += 32 * 16;
    __builtin_prefetch(bp, 0, 1);
    v16bf a0 = make_afrag(a0r + k0 + 8 * hi, a0r + k0 + 16 + 8 * hi);
    v16bf a1 = make_afrag(a1r + k0 + 8 * hi, a1r + k0 + 16 + 8 * hi);
    c0 = wmma_bf16(a0, bfrag, c0);
    c1 = wmma_bf16(a1, bfrag, c1);
  }

  int col = nt * 16 + m0;                           // h-unit index (N == H)
  float bdv = bd[col];
  #pragma unroll
  for (int r = 0; r < 8; ++r) {
    #pragma unroll
    for (int half = 0; half < 2; ++half) {
      int    row  = r + 8 * hi + 16 * half;
      float  qv   = tanhf((half ? c1[r] : c0[r]) + bdv);
      size_t base = (size_t)row * MH + (size_t)col * M_;
      float  hn   = 0.f;
      #pragma unroll
      for (int m = 0; m < M_; ++m) {
        float sk = ski[base + m];
        float et = ecol[row * M_ + m];
        float nv = ((1.f - sk) * hhat[base + m] + sk * qv) * et;
        hhat[base + m]  = nv;
        hflat[base + m] = tobf(nv);
        hn += nv;
      }
      fused[row * IH + I_ + col] = tobf(hn);
    }
  }
}

// ---------------------------------------------------------------------------
// Decode: out = hflat[32,8192] @ Wdec + bdec -> [32, 512].  32 waves -> <<<4,256>>>
// ---------------------------------------------------------------------------
__global__ void decode_gemm(const __bf16* __restrict__ hflat, const __bf16* __restrict__ Wdp,
                            const float* __restrict__ bdec, float* __restrict__ out) {
  int gtid = blockIdx.x * blockDim.x + threadIdx.x;
  int wave = gtid >> 5;                             // 0..31 = n-tile
  int lane = gtid & 31;
  int nt = wave;
  int m0 = lane & 15;
  int hi = lane >> 4;
  const __bf16* a0r = hflat + (size_t)m0 * MH;
  const __bf16* a1r = hflat + (size_t)(m0 + 16) * MH;
  const __bf16* bp  = Wdp + ((size_t)(nt * 256) * 32 + lane) * 16;

  v8f c0 = {0.f,0.f,0.f,0.f,0.f,0.f,0.f,0.f};
  v8f c1 = {0.f,0.f,0.f,0.f,0.f,0.f,0.f,0.f};
  for (int kt = 0; kt < 256; ++kt) {
    int k0 = kt * 32;
    v16bf bfrag = *(const v16bf*)bp;
    bp += 32 * 16;
    __builtin_prefetch(bp, 0, 1);
    v16bf a0 = make_afrag(a0r + k0 + 8 * hi, a0r + k0 + 16 + 8 * hi);
    v16bf a1 = make_afrag(a1r + k0 + 8 * hi, a1r + k0 + 16 + 8 * hi);
    c0 = wmma_bf16(a0, bfrag, c0);
    c1 = wmma_bf16(a1, bfrag, c1);
  }
  int col = nt * 16 + m0;
  float bv = bdec[col];
  #pragma unroll
  for (int r = 0; r < 8; ++r) {
    int row = r + 8 * hi;
    out[(size_t)row        * O_ + col] = c0[r] + bv;
    out[(size_t)(row + 16) * O_ + col] = c1[r] + bv;
  }
}

// ---------------------------------------------------------------------------
extern "C" void kernel_launch(void* const* d_in, const int* in_sizes, int n_in,
                              void* d_out, int out_size, void* d_ws, size_t ws_size,
                              hipStream_t stream) {
  const float* t    = (const float*)d_in[0];
  const float* x    = (const float*)d_in[1];
  const float* Wr   = (const float*)d_in[2];
  const float* br   = (const float*)d_in[3];
  const float* Wd   = (const float*)d_in[4];
  const float* bd   = (const float*)d_in[5];
  const float* Wu   = (const float*)d_in[6];
  const float* bu   = (const float*)d_in[7];
  const float* Wdec = (const float*)d_in[8];
  const float* bdec = (const float*)d_in[9];
  float* out = (float*)d_out;

  char* ws = (char*)d_ws;
  size_t off = 0;
  auto alloc = [&](size_t bytes) -> void* {
    void* p = ws + off;
    off = (off + bytes + 255) & ~(size_t)255;
    return p;
  };
  __bf16* Wrp   = (__bf16*)alloc((size_t)IH * MH * 2);      // 21.0 MB
  __bf16* Wup   = (__bf16*)alloc((size_t)IH * MH * 2);      // 21.0 MB
  __bf16* Wdp   = (__bf16*)alloc((size_t)IH * H_ * 2);      //  2.6 MB
  __bf16* Wdecp = (__bf16*)alloc((size_t)MH * O_ * 2);      //  8.4 MB
  float*  hhat  = (float*) alloc((size_t)B_ * MH * 4);      //  1.0 MB
  float*  skib  = (float*) alloc((size_t)B_ * MH * 4);      //  1.0 MB
  __bf16* fused = (__bf16*)alloc((size_t)B_ * IH * 2);      //  80 KB
  __bf16* xq    = (__bf16*)alloc((size_t)B_ * IH * 2);      //  80 KB
  __bf16* hflat = (__bf16*)alloc((size_t)B_ * MH * 2);      // 512 KB
  (void)ws_size; (void)in_sizes; (void)n_in; (void)out_size;

  // Per-launch weight packing fp32 -> B-fragment-tiled bf16
  pack_weight<<<(IH * MH / 2 + 255) / 256, 256, 0, stream>>>(Wr,   (unsigned*)Wrp,   IH, MH, 40);
  pack_weight<<<(IH * MH / 2 + 255) / 256, 256, 0, stream>>>(Wu,   (unsigned*)Wup,   IH, MH, 40);
  pack_weight<<<(IH * H_ / 2 + 255) / 256, 256, 0, stream>>>(Wd,   (unsigned*)Wdp,   IH, H_, 40);
  pack_weight<<<(MH * O_ / 2 + 255) / 256, 256, 0, stream>>>(Wdec, (unsigned*)Wdecp, MH, O_, 256);

  init_state<<<(B_ * MH + 255) / 256, 256, 0, stream>>>(x, hhat, fused);

  for (int step = 0; step < T_; ++step) {
    gates_gemm<<<128, 256, 0, stream>>>(fused, Wrp, Wup, br, bu, hhat, x, step, xq, skib);
    decay_gemm<<<8, 256, 0, stream>>>(xq, Wdp, bd, skib, t, x, step, hhat, fused, hflat);
  }
  decode_gemm<<<4, 256, 0, stream>>>(hflat, Wdecp, bdec, out);
}
